// BF16AccumLinear_70171175682829
// MI455X (gfx1250) — matrix operand order repaired and checked by reading, not compiled, not run on hardware
//
#include <hip/hip_runtime.h>
#include <hip/hip_bf16.h>

// ---- CDNA5 WMMA types ----
typedef __attribute__((ext_vector_type(16))) __bf16 v16bf;
typedef __attribute__((ext_vector_type(8)))  __bf16 v8bf;
typedef __attribute__((ext_vector_type(8)))  float  v8f;

// GEMM: y = x @ W^T + b ; M=16384, N=1024, K=1024, bf16 in / bf16 out
// Block tile 128x128, K-step 32, 256 threads (8 waves), wave tile 32x64.
// 3-stage async-DMA ring (ASYNCcnt <= 4 lookahead), LDS-restaged epilogue.
constexpr int DIM_K     = 1024;
constexpr int DIM_N     = 1024;
constexpr int TILE_K    = 32;
constexpr int LDS_PITCH = 40;               // 32 elems + 8 pad -> 80 B/row
constexpr int NK        = DIM_K / TILE_K;   // 32
constexpr unsigned A_BYTES      = 128 * LDS_PITCH * 2;      // 10240
constexpr unsigned STAGE_BYTES  = 2 * A_BYTES;              // A + B per stage
constexpr unsigned SMEM_BYTES   = 3 * STAGE_BYTES;          // 61440 (> C restage 34816)
constexpr int C_PITCH  = 136;               // epilogue restage pitch (272 B, 16B-aligned)

__device__ __forceinline__ void async_copy_b128(unsigned lds_off, const void* gaddr) {
    asm volatile("global_load_async_to_lds_b128 %0, %1, off"
                 :: "v"(lds_off), "v"(gaddr) : "memory");
}
__device__ __forceinline__ void wait_async_le4() {
#if __has_builtin(__builtin_amdgcn_s_wait_asynccnt)
    __builtin_amdgcn_s_wait_asynccnt(4);
#else
    asm volatile("s_wait_asynccnt 0x4" ::: "memory");
#endif
}
__device__ __forceinline__ void wait_async0() {
#if __has_builtin(__builtin_amdgcn_s_wait_asynccnt)
    __builtin_amdgcn_s_wait_asynccnt(0);
#else
    asm volatile("s_wait_asynccnt 0x0" ::: "memory");
#endif
}

__global__ __launch_bounds__(256) void bf16_linear_wmma_kernel(
    const unsigned short* __restrict__ A,   // x (M,K) row-major bf16 bits
    const unsigned short* __restrict__ W,   // weight (N,K) row-major bf16 bits
    const __hip_bfloat16* __restrict__ bias,
    __hip_bfloat16*       __restrict__ out) // (M,N) bf16
{
    __shared__ __align__(16) unsigned char smem[SMEM_BYTES];

    const int t       = threadIdx.x;
    const int lane    = t & 31;
    const int wv      = t >> 5;
    const int waveM   = (wv >> 1) * 32;     // 0,32,64,96
    const int waveN   = (wv & 1) * 64;      // 0,64
    const int r       = lane & 15;
    const bool hiHalf = (lane >= 16);

    const int blockN = blockIdx.x * 128;
    const int blockM = blockIdx.y * 128;

    // global->LDS slab mapping: 256 threads cover 128 rows x 64 B in 2 passes
    const int row0 = t >> 2;                // 0..63
    const int colE = (t & 3) * 8;           // 0,8,16,24 elems (16B chunks)

    const unsigned short* gA0 = A + (size_t)(blockM + row0)      * DIM_K + colE;
    const unsigned short* gA1 = A + (size_t)(blockM + row0 + 64) * DIM_K + colE;
    const unsigned short* gB0 = W + (size_t)(blockN + row0)      * DIM_K + colE;
    const unsigned short* gB1 = W + (size_t)(blockN + row0 + 64) * DIM_K + colE;

    // per-thread LDS byte offsets (generic-shared low 32 bits = LDS offset)
    const unsigned base = (unsigned)(size_t)&smem[0];
    const unsigned pa0  = base + (row0 * LDS_PITCH + colE) * 2;
    const unsigned pa1  = pa0 + 64 * LDS_PITCH * 2;
    const unsigned pb0  = pa0 + A_BYTES;
    const unsigned pb1  = pb0 + 64 * LDS_PITCH * 2;

    auto issue_stage = [&](int kt, int st) {
        const int k0 = kt * TILE_K;
        const unsigned bs = (unsigned)st * STAGE_BYTES;
        async_copy_b128(pa0 + bs, gA0 + k0);
        async_copy_b128(pa1 + bs, gA1 + k0);
        async_copy_b128(pb0 + bs, gB0 + k0);
        async_copy_b128(pb1 + bs, gB1 + k0);
    };

    // accumulators: 2 (M) x 4 (N) tiles of 16x16 f32
    v8f acc[2][4];
    #pragma unroll
    for (int mt = 0; mt < 2; ++mt)
        #pragma unroll
        for (int nt = 0; nt < 4; ++nt)
            acc[mt][nt] = v8f{};

    // A-frag (ISA 16-bit A 16x32): lanes 0-15 K 0-7|16-23, lanes 16-31 K 8-15|24-31
    const int koff   = hiHalf ? 8 : 0;
    const int kstart = hiHalf ? 16 : 0;     // B-frag: lane n = column N, same K split as A-frag (koff)

    auto compute_stage = [&](int st) {
        const unsigned short* Ab = (const unsigned short*)(smem + (unsigned)st * STAGE_BYTES);
        const unsigned short* Bb = (const unsigned short*)(smem + (unsigned)st * STAGE_BYTES + A_BYTES);

        v16bf afrag[2];
        #pragma unroll
        for (int mt = 0; mt < 2; ++mt) {
            const unsigned short* p = Ab + (waveM + mt * 16 + r) * LDS_PITCH;
            v8bf lo = *(const v8bf*)(p + koff);
            v8bf hi = *(const v8bf*)(p + 16 + koff);
            afrag[mt] = __builtin_shufflevector(lo, hi,
                0,1,2,3,4,5,6,7,8,9,10,11,12,13,14,15);
        }
        v16bf bfrag[4];
        #pragma unroll
        for (int nt = 0; nt < 4; ++nt) {
            const unsigned short* p = Bb + (waveN + nt * 16 + r) * LDS_PITCH + koff;
            v8bf lo = *(const v8bf*)(p);
            v8bf hi = *(const v8bf*)(p + 16);
            bfrag[nt] = __builtin_shufflevector(lo, hi,
                0,1,2,3,4,5,6,7,8,9,10,11,12,13,14,15);
        }
        #pragma unroll
        for (int mt = 0; mt < 2; ++mt)
            #pragma unroll
            for (int nt = 0; nt < 4; ++nt)
                acc[mt][nt] = __builtin_amdgcn_wmma_f32_16x16x32_bf16(
                    false, afrag[mt], false, bfrag[nt],
                    (short)0, acc[mt][nt], false, false);
    };

    // ---- 3-stage software pipeline ----
    issue_stage(0, 0);
    issue_stage(1, 1);

    int cur = 0, stIssue = 2;
    for (int kt = 0; kt < NK - 1; ++kt) {
        wait_async_le4();                   // stage kt landed (in-order retirement)
        __syncthreads();                    // ... for every wave in the group
        if (kt + 2 < NK) {
            issue_stage(kt + 2, stIssue);   // readers of that buffer passed barrier
            stIssue = (stIssue == 2) ? 0 : stIssue + 1;
        }
        compute_stage(cur);
        cur = (cur == 2) ? 0 : cur + 1;
    }
    wait_async0();                          // peeled last iteration
    __syncthreads();
    compute_stage(cur);

    // ---- epilogue: restage C through LDS for coalesced b128 stores ----
    __syncthreads();                        // stage buffers dead; reuse LDS
    unsigned short* ctile = (unsigned short*)smem;   // 128 x C_PITCH bf16

    #pragma unroll
    for (int nt = 0; nt < 4; ++nt) {
        const int cl   = waveN + nt * 16 + r;          // local column
        const float bv = __bfloat162float(bias[blockN + cl]);
        #pragma unroll
        for (int mt = 0; mt < 2; ++mt) {
            const int rl = waveM + mt * 16 + (hiHalf ? 8 : 0);
            unsigned short* p = ctile + rl * C_PITCH + cl;
            #pragma unroll
            for (int v = 0; v < 8; ++v) {
                __hip_bfloat16 h = __float2bfloat16(acc[mt][nt][v] + bv);
                p[v * C_PITCH] = __builtin_bit_cast(unsigned short, h);
            }
        }
    }
    __syncthreads();

    // coalesced write-out: thread -> 128 contiguous bytes of one output row
    const int orow = t >> 1;                // 0..127
    const int ocol = (t & 1) * 64;          // elems
    const unsigned short* src = ctile + orow * C_PITCH + ocol;
    __hip_bfloat16* dst = out + (size_t)(blockM + orow) * DIM_N + blockN + ocol;
    #pragma unroll
    for (int i = 0; i < 8; ++i)
        *(float4*)((unsigned short*)dst + i * 8) = *(const float4*)(src + i * 8);
}

extern "C" void kernel_launch(void* const* d_in, const int* in_sizes, int n_in,
                              void* d_out, int out_size, void* d_ws, size_t ws_size,
                              hipStream_t stream) {
    (void)n_in; (void)d_ws; (void)ws_size; (void)out_size;

    const unsigned short* x = (const unsigned short*)d_in[0];   // bf16 bits
    const unsigned short* w = (const unsigned short*)d_in[1];   // bf16 bits
    const __hip_bfloat16* b = (const __hip_bfloat16*)d_in[2];
    __hip_bfloat16* out = (__hip_bfloat16*)d_out;

    const int M = in_sizes[0] / DIM_K;      // 16384

    dim3 grid(DIM_N / 128, M / 128);        // (8, 128)
    dim3 block(256);
    bf16_linear_wmma_kernel<<<grid, block, 0, stream>>>(x, w, b, out);
}
